// MPNN_6373731467378
// MI455X (gfx1250) — compile-verified, matrix operand order checked by reading, not run on hardware
//
#include <hip/hip_runtime.h>

// ---------------------------------------------------------------------------
// MPNN forward for MI455X (gfx1250), wave32, WMMA bf16 path.
//
// d_in index map (assumption: top-level dict insertion order, nested params
// flattened with dict keys sorted alphabetically, lists in order):
//   0 X(4,2048,8,4) 1 E(4,2048,2048,2) 2 y(4,16) 3 label(4,2048) i32 4 node_mask(4,2048)
//   gnn: 5 emb(11,32) 6..9 in_X[0].{W,b} in_X[1].{W,b} 10..13 in_y[...]
//        14+12*l per layer: gcn_X.W,gcn_X.b,gcn_l.W,gcn_l.b,ln_X.b,ln_X.g,
//                           ln_l.b,ln_l.g,up_X.W,up_X.b,up_l.W,up_l.b   (l=0,1)
//        38..41 out1[0..1].{W,b}  42..45 out2[0..1].{W,b}
//   mlp: 46 emb(11,64) 47..50 in_X 51..54 in_y
//        55+8*l per layer: ln_X.b,ln_X.g,ln_l.b,ln_l.g,up_X.W,up_X.b,up_l.W,up_l.b
//        71..74 out[0..1].{W,b}
// d_out: [Xm (4*2048*32)] [Em (4*2048*2)] [y (64)]  = 278592 floats.
// ---------------------------------------------------------------------------

typedef __attribute__((ext_vector_type(16))) __bf16 v16bf;
typedef __attribute__((ext_vector_type(8)))  __bf16 v8bf;
typedef __attribute__((ext_vector_type(4)))  __bf16 v4bf;
typedef __attribute__((ext_vector_type(8)))  float  v8f;

#define TM 128   // block tile rows   (8 waves x 16)
#define TN 32    // block tile cols   (2 wmma tiles per wave)
#define TK 32    // K step (one bf16 wmma)
#define LDS_STRIDE 40  // bf16 elems per LDS row (80B: 16B-aligned rows, staggers banks)

template<typename T>
__device__ __forceinline__ float ldAf(const T* p) { return (float)(*p); }

// C[z] = epilogue(A[z] @ B[z] + bias), A:(M,K) row-major (fp32 or bf16),
// B:(K,Nc) row-major fp32, C:(M,Nc) fp32. epilogue: +bias[col], relu, *rowscale[row].
// Interior tiles use block-uniform predicates (float4/b64 staging, two
// ds_load_b128 per fragment). LDS is double-buffered: stage tile k+1 while
// computing tile k -> one workgroup barrier per K-step.
template<bool RELU, typename TA>
__global__ __launch_bounds__(256)
void wgemm(const TA* __restrict__ A, const float* __restrict__ B,
           const float* __restrict__ bias, const float* __restrict__ rowscale,
           float* __restrict__ C, int M, int Nc, int K,
           long sA, long sB, long sC)
{
    __shared__ __align__(16) __bf16 As [2][TM][LDS_STRIDE];
    __shared__ __align__(16) __bf16 BsT[2][TN][LDS_STRIDE];   // transposed: [n][k]

    const int  tid  = threadIdx.x;
    const int  lane = tid & 31;
    const int  wv   = tid >> 5;                 // wave 0..7 -> rows wv*16..+15
    const int  half = lane >> 4;                // K-half selector (wave32 layout)
    const int  l15  = lane & 15;
    const long m0   = (long)blockIdx.y * TM;
    const long n0   = (long)blockIdx.x * TN;

    const TA*    Ab = A + (long)blockIdx.z * sA;
    const float* Bb = B + (long)blockIdx.z * sB;
    float*       Cb = C + (long)blockIdx.z * sC;

    const bool fullM = (m0 + TM <= M);          // block-uniform
    const bool fullN = (n0 + TN <= Nc);

    const int rS = tid >> 3;                    // staging row helper (0..31)
    const int cS = (tid & 7) * 4;               // staging col helper (0,4,..,28)

    // stage global tile at k0 into LDS buffer `buf`
    auto stage = [&](int buf, int k0) {
        const bool fullK = (k0 + TK <= K);      // block-uniform
        // ---- A tile (128x32) -> LDS bf16 ----
        if (fullM && fullK) {
            #pragma unroll
            for (int rr = 0; rr < TM; rr += 32) {
                const int  r   = rr + rS;
                const TA*  src = Ab + (m0 + r) * (long)K + k0 + cS;
                if (k0 + TK < K)                 // uniform: prefetch tile after next
                    __builtin_prefetch(src + TK, 0, 1);
                v4bf w;
                if constexpr (sizeof(TA) == 4) {
                    const float4 v = *(const float4*)src;   // global_load_b128
                    w[0] = (__bf16)v.x; w[1] = (__bf16)v.y;
                    w[2] = (__bf16)v.z; w[3] = (__bf16)v.w;
                } else {
                    w = *(const v4bf*)src;                  // global_load_b64 (bf16 A)
                }
                *(v4bf*)&As[buf][r][cS] = w;                // ds_store_b64
            }
        } else {
            #pragma unroll
            for (int rr = 0; rr < TM; rr += 32) {
                const int  r  = rr + rS;
                const long gr = m0 + r;
                #pragma unroll
                for (int u = 0; u < 4; ++u) {
                    const int c = cS + u, gk = k0 + c;
                    float v = (gr < M && gk < K) ? ldAf(Ab + gr * (long)K + gk) : 0.0f;
                    As[buf][r][c] = (__bf16)v;
                }
            }
        }
        // ---- B tile (32x32) transposed -> LDS bf16 ----
        if (fullN && fullK) {
            const float4 v = *(const float4*)&Bb[(long)(k0 + rS) * Nc + n0 + cS];
            BsT[buf][cS + 0][rS] = (__bf16)v.x;
            BsT[buf][cS + 1][rS] = (__bf16)v.y;
            BsT[buf][cS + 2][rS] = (__bf16)v.z;
            BsT[buf][cS + 3][rS] = (__bf16)v.w;
        } else {
            const int gk = k0 + rS;
            #pragma unroll
            for (int u = 0; u < 4; ++u) {
                const long gc = n0 + cS + u;
                float v = (gk < K && gc < Nc) ? Bb[(long)gk * Nc + gc] : 0.0f;
                BsT[buf][cS + u][rS] = (__bf16)v;
            }
        }
    };

    v8f acc0 = {}; v8f acc1 = {};

    stage(0, 0);
    __syncthreads();

    int buf = 0;
    for (int k0 = 0; k0 < K; k0 += TK) {
        const int nxt = buf ^ 1;
        if (k0 + TK < K) stage(nxt, k0 + TK);   // overlap next-tile staging w/ compute

        // ---- fragment loads: two contiguous 8xbf16 runs => 2x ds_load_b128 ----
        const int arow = wv * 16 + l15;
        const v8bf alo = *(const v8bf*)&As[buf][arow][half * 8];
        const v8bf ahi = *(const v8bf*)&As[buf][arow][16 + half * 8];
        const v16bf af = __builtin_shufflevector(alo, ahi,
                           0,1,2,3,4,5,6,7,8,9,10,11,12,13,14,15);

        const v8bf b0lo = *(const v8bf*)&BsT[buf][l15][half * 8];
        const v8bf b0hi = *(const v8bf*)&BsT[buf][l15][16 + half * 8];
        const v16bf bf0 = __builtin_shufflevector(b0lo, b0hi,
                           0,1,2,3,4,5,6,7,8,9,10,11,12,13,14,15);

        const v8bf b1lo = *(const v8bf*)&BsT[buf][l15 + 16][half * 8];
        const v8bf b1hi = *(const v8bf*)&BsT[buf][l15 + 16][16 + half * 8];
        const v16bf bf1 = __builtin_shufflevector(b1lo, b1hi,
                           0,1,2,3,4,5,6,7,8,9,10,11,12,13,14,15);

        acc0 = __builtin_amdgcn_wmma_f32_16x16x32_bf16(false, af, false, bf0,
                                                       (short)0, acc0, false, false);
        acc1 = __builtin_amdgcn_wmma_f32_16x16x32_bf16(false, af, false, bf1,
                                                       (short)0, acc1, false, false);
        __syncthreads();                        // one barrier per K-step
        buf = nxt;
    }

    // ---- epilogue: D layout -> m = r + 8*half, n = l15 (+16 second tile) ----
    const long gc0 = n0 + l15;
    const long gc1 = n0 + 16 + l15;
    if (fullM && fullN) {
        const float bia0 = bias ? bias[gc0] : 0.0f;
        const float bia1 = bias ? bias[gc1] : 0.0f;
        #pragma unroll
        for (int r = 0; r < 8; ++r) {
            const long gm = m0 + wv * 16 + r + half * 8;
            const float rs = rowscale ? rowscale[gm] : 1.0f;
            float v0 = acc0[r] + bia0;
            float v1 = acc1[r] + bia1;
            if (RELU) { v0 = fmaxf(v0, 0.0f); v1 = fmaxf(v1, 0.0f); }
            Cb[gm * (long)Nc + gc0] = v0 * rs;
            Cb[gm * (long)Nc + gc1] = v1 * rs;
        }
    } else {
        #pragma unroll
        for (int r = 0; r < 8; ++r) {
            const long gm = m0 + wv * 16 + r + half * 8;
            if (gm < M) {
                const float rs = rowscale ? rowscale[gm] : 1.0f;
                if (gc0 < Nc) {
                    float v = acc0[r];
                    if (bias) v += bias[gc0];
                    if (RELU) v = fmaxf(v, 0.0f);
                    Cb[gm * (long)Nc + gc0] = v * rs;
                }
                if (gc1 < Nc) {
                    float v = acc1[r];
                    if (bias) v += bias[gc1];
                    if (RELU) v = fmaxf(v, 0.0f);
                    Cb[gm * (long)Nc + gc1] = v * rs;
                }
            }
        }
    }
}

// out = LN(relu(x)) * gamma + beta, * mask[row]. One wave per row (wave32).
__global__ __launch_bounds__(256)
void ln_relu_mask(const float* __restrict__ x, const float* __restrict__ gamma,
                  const float* __restrict__ beta, const float* __restrict__ mask,
                  float* __restrict__ out, int M, int C)
{
    const int  warp = threadIdx.x >> 5;
    const int  lane = threadIdx.x & 31;
    const long row  = (long)blockIdx.x * 8 + warp;
    if (row >= M) return;
    const float* xr = x + row * (long)C;
    const int per = C >> 5;                       // C in {32,64,128,256}
    float vals[8];
    float s = 0.f;
    for (int i = 0; i < per; ++i) {
        float v = fmaxf(xr[lane + (i << 5)], 0.f);
        vals[i] = v; s += v;
    }
    for (int o = 16; o > 0; o >>= 1) s += __shfl_xor(s, o, 32);
    const float mu = s / (float)C;
    float vs = 0.f;
    for (int i = 0; i < per; ++i) { float d = vals[i] - mu; vs += d * d; }
    for (int o = 16; o > 0; o >>= 1) vs += __shfl_xor(vs, o, 32);
    const float inv  = rsqrtf(vs / (float)C + 1e-5f);
    const float mrow = mask[row];
    for (int i = 0; i < per; ++i) {
        const int c = lane + (i << 5);
        out[row * (long)C + c] = ((vals[i] - mu) * inv * gamma[c] + beta[c]) * mrow;
    }
}

// dinv[b,i] = mask[b,i]>0 ? rsqrt(1 + sum_j adj[b,i,j]*mask[b,j]) : 0
__global__ __launch_bounds__(256)
void k_dinv(const float* __restrict__ E, const float* __restrict__ mask,
            float* __restrict__ dinv, int N)
{
    const long row = blockIdx.x;                  // b*N + i
    const long b   = row / N;
    const float* mb = mask + b * N;
    const float* er = E + row * (long)N * 2;      // E[b,i,:,1] at stride 2
    float s = 0.f;
    for (int j = threadIdx.x; j < N; j += 256) s += er[2 * j + 1] * mb[j];
    __shared__ float red[256];
    red[threadIdx.x] = s; __syncthreads();
    for (int st = 128; st > 0; st >>= 1) {
        if (threadIdx.x < st) red[threadIdx.x] += red[threadIdx.x + st];
        __syncthreads();
    }
    if (threadIdx.x == 0)
        dinv[row] = (mask[row] > 0.f) ? rsqrtf(red[0] + 1.0f) : 0.f;
}

// Anorm[b,i,j] = (adj + (i==j)) * dinv[b,i] * dinv[b,j], stored bf16 (dinv encodes mask)
__global__ void k_anorm(const float* __restrict__ E, const float* __restrict__ dinv,
                        __bf16* __restrict__ A, int N, long total)
{
    const long i = (long)blockIdx.x * blockDim.x + threadIdx.x;
    if (i >= total) return;
    const long j  = i % N;
    const long bi = i / N;
    const long ii = bi % N;
    const long b  = bi / N;
    float a = E[i * 2 + 1];
    if (ii == j) a += 1.0f;                       // adj diag is 0 by construction
    A[i] = (__bf16)(a * dinv[bi] * dinv[b * (long)N + j]);
}

__global__ void k_copy_cols(const float* __restrict__ s, float* __restrict__ d,
                            long total, int srcC, int dstC, int off)
{
    const long i = (long)blockIdx.x * blockDim.x + threadIdx.x;
    if (i >= total) return;
    const long r = i / srcC;
    const int  c = (int)(i - r * srcC);
    d[r * (long)dstC + off + c] = s[i];
}

__global__ void k_embed(const float* __restrict__ emb, const int* __restrict__ label,
                        const float* __restrict__ mask, float* __restrict__ o,
                        long total, int C)
{
    const long i = (long)blockIdx.x * blockDim.x + threadIdx.x;
    if (i >= total) return;
    const long r = i / C;
    const int  c = (int)(i - r * C);
    o[i] = emb[(long)(label[r] + 1) * C + c] * mask[r];
}

__global__ void k_bcast(const float* __restrict__ yh, float* __restrict__ o,
                        long total, int C, int N)
{
    const long i = (long)blockIdx.x * blockDim.x + threadIdx.x;
    if (i >= total) return;
    const long r = i / C;
    const int  c = (int)(i - r * C);
    o[i] = yh[(r / N) * (long)C + c];
}

__global__ void k_scale_rows(const float* __restrict__ s, const float* __restrict__ mask,
                             float* __restrict__ o, long total, int C)
{
    const long i = (long)blockIdx.x * blockDim.x + threadIdx.x;
    if (i >= total) return;
    o[i] = s[i] * mask[i / C];
}

__global__ void k_copy(const float* __restrict__ s, float* __restrict__ d, int n)
{
    const int i = blockIdx.x * blockDim.x + threadIdx.x;
    if (i < n) d[i] = s[i];
}

// ---------------------------------------------------------------------------
extern "C" void kernel_launch(void* const* d_in, const int* in_sizes, int n_in,
                              void* d_out, int out_size, void* d_ws, size_t ws_size,
                              hipStream_t stream)
{
    (void)in_sizes; (void)n_in; (void)out_size; (void)ws_size;
    const int BSz = 4, Nn = 2048, Mr = BSz * Nn;   // 8192 node-rows

    const float* X     = (const float*)d_in[0];
    const float* E     = (const float*)d_in[1];
    const float* Y     = (const float*)d_in[2];
    const int*   label = (const int*)  d_in[3];
    const float* mask  = (const float*)d_in[4];
    auto P = [&](int i) { return (const float*)d_in[i]; };
    float* out = (float*)d_out;

    // ---- workspace carve-out (bump allocator, 256B aligned) ----
    char* wp = (char*)d_ws;
    auto aF = [&](size_t n) -> float* {
        float* p = (float*)wp; wp += (n * sizeof(float) + 255) & ~(size_t)255; return p; };
    auto aH = [&](size_t n) -> __bf16* {
        __bf16* p = (__bf16*)wp; wp += (n * sizeof(__bf16) + 255) & ~(size_t)255; return p; };

    __bf16* Anorm = aH((size_t)BSz * Nn * Nn);     // 33.5 MB bf16 (dominant stream)
    float* dinv   = aF(Mr);
    // MLP buffers
    float* m_t0   = aF((size_t)Mr * 256);
    float* m_h    = aF((size_t)Mr * 256);
    float* m_y0   = aF(4 * 64);
    float* m_yh   = aF(4 * 64);
    float* m_ye   = aF((size_t)Mr * 64);
    float* m_lab  = aF((size_t)Mr * 64);
    float* m_cat  = aF((size_t)Mr * 384);
    float* m_pre  = aF((size_t)Mr * 256);
    float* m_lpre = aF((size_t)Mr * 64);
    float* hcat_m = aF((size_t)Mr * 1024);
    float* m_big  = aF((size_t)Mr * 1024);
    float* Xout   = aF((size_t)Mr * 32);
    // GNN buffers
    float* g_t0   = aF((size_t)Mr * 128);
    float* g_h    = aF((size_t)Mr * 128);
    float* g_y0   = aF(4 * 32);
    float* g_yh   = aF(4 * 32);
    float* g_ye   = aF((size_t)Mr * 32);
    float* g_lab  = aF((size_t)Mr * 32);
    float* g_cat0 = aF((size_t)Mr * 160);
    float* g_t128 = aF((size_t)Mr * 128);
    float* g_hx   = aF((size_t)Mr * 128);
    float* g_t32  = aF((size_t)Mr * 32);
    float* g_lb   = aF((size_t)Mr * 32);
    float* g_cat2 = aF((size_t)Mr * 192);
    float* g_pre  = aF((size_t)Mr * 128);
    float* g_lpre = aF((size_t)Mr * 32);
    float* hcat_g = aF((size_t)Mr * 512);
    float* g_eh1  = aF((size_t)Mr * 512);
    float* g_eh   = aF((size_t)Mr * 128);
    float* g_e2   = aF((size_t)Mr * 128);

    auto gemmF = [&](const float* A, const float* B, const float* bias, const float* rs,
                     float* C, int M, int Nc, int K, bool relu) {
        dim3 g((Nc + TN - 1) / TN, (M + TM - 1) / TM, 1);
        if (relu) wgemm<true , float><<<g, 256, 0, stream>>>(A, B, bias, rs, C, M, Nc, K, 0, 0, 0);
        else      wgemm<false, float><<<g, 256, 0, stream>>>(A, B, bias, rs, C, M, Nc, K, 0, 0, 0);
    };
    auto gemmAgg = [&](const __bf16* A, const float* B, const float* bias, float* C, int Nc) {
        dim3 g((Nc + TN - 1) / TN, (Nn + TM - 1) / TM, BSz);
        wgemm<false, __bf16><<<g, 256, 0, stream>>>(A, B, bias, nullptr, C, Nn, Nc, Nn,
                                                    (long)Nn * Nn, (long)Nn * Nc, (long)Nn * Nc);
    };
    auto g1 = [&](long tot) { return dim3((unsigned)((tot + 255) / 256)); };
    const dim3 lnGrid((Mr + 7) / 8);

    // ---------- GCN normalization: D^{-1/2}(A+I)D^{-1/2}, bf16 ----------
    k_dinv<<<dim3(Mr), 256, 0, stream>>>(E, mask, dinv, Nn);
    { long tot = (long)BSz * Nn * Nn;
      k_anorm<<<g1(tot), 256, 0, stream>>>(E, dinv, Anorm, Nn, tot); }

    // ---------- attr predictor (MLP) ----------
    gemmF(X,    P(47), P(48), nullptr, m_t0, Mr, 256, 32,  true);
    gemmF(m_t0, P(49), P(50), mask,    m_h,  Mr, 256, 256, true);   // h = mlp2(X)*mask
    gemmF(Y,    P(51), P(52), nullptr, m_y0, 4, 64, 16, true);
    gemmF(m_y0, P(53), P(54), nullptr, m_yh, 4, 64, 64, true);      // yh
    k_embed<<<g1((long)Mr * 64), 256, 0, stream>>>(P(46), label, mask, m_lab, (long)Mr * 64, 64);
    k_bcast<<<g1((long)Mr * 64), 256, 0, stream>>>(m_yh, m_ye, (long)Mr * 64, 64, Nn);
    k_copy_cols<<<g1((long)Mr * 256), 256, 0, stream>>>(m_h,   hcat_m, (long)Mr * 256, 256, 1024, 0);
    k_copy_cols<<<g1((long)Mr * 64 ), 256, 0, stream>>>(m_lab, hcat_m, (long)Mr * 64,  64,  1024, 768);
    for (int l = 0; l < 2; ++l) {
        const int b = 55 + 8 * l;  // ln_X.b, ln_X.g, ln_l.b, ln_l.g, up_X.W, up_X.b, up_l.W, up_l.b
        k_copy_cols<<<g1((long)Mr * 256), 256, 0, stream>>>(m_h,   m_cat, (long)Mr * 256, 256, 384, 0);
        k_copy_cols<<<g1((long)Mr * 64 ), 256, 0, stream>>>(m_lab, m_cat, (long)Mr * 64,  64,  384, 256);
        k_copy_cols<<<g1((long)Mr * 64 ), 256, 0, stream>>>(m_ye,  m_cat, (long)Mr * 64,  64,  384, 320);
        gemmF(m_cat, P(b + 4), P(b + 5), nullptr, m_pre, Mr, 256, 384, false);
        ln_relu_mask<<<lnGrid, 256, 0, stream>>>(m_pre, P(b + 1), P(b + 0), mask, m_h, Mr, 256);
        gemmF(m_lab, P(b + 6), P(b + 7), nullptr, m_lpre, Mr, 64, 64, false);
        ln_relu_mask<<<lnGrid, 256, 0, stream>>>(m_lpre, P(b + 3), P(b + 2), mask, m_lab, Mr, 64);
        k_copy_cols<<<g1((long)Mr * 256), 256, 0, stream>>>(m_h,   hcat_m, (long)Mr * 256, 256, 1024, 256 * (l + 1));
        k_copy_cols<<<g1((long)Mr * 64 ), 256, 0, stream>>>(m_lab, hcat_m, (long)Mr * 64,  64,  1024, 768 + 64 * (l + 1));
    }
    k_copy_cols<<<g1((long)Mr * 64), 256, 0, stream>>>(m_ye, hcat_m, (long)Mr * 64, 64, 1024, 960);
    gemmF(hcat_m, P(71), P(72), nullptr, m_big, Mr, 1024, 1024, true);
    gemmF(m_big,  P(73), P(74), nullptr, Xout,  Mr, 32,   1024, false);
    k_scale_rows<<<g1((long)Mr * 32), 256, 0, stream>>>(Xout, mask, out, (long)Mr * 32, 32);

    // ---------- link predictor (GNN) ----------
    gemmF(Xout, P(6), P(7), nullptr, g_t0, Mr, 128, 32,  true);
    gemmF(g_t0, P(8), P(9), mask,    g_h,  Mr, 128, 128, true);
    gemmF(Y,    P(10), P(11), nullptr, g_y0, 4, 32, 16, true);
    gemmF(g_y0, P(12), P(13), nullptr, g_yh, 4, 32, 32, true);
    k_embed<<<g1((long)Mr * 32), 256, 0, stream>>>(P(5), label, mask, g_lab, (long)Mr * 32, 32);
    k_bcast<<<g1((long)Mr * 32), 256, 0, stream>>>(g_yh, g_ye, (long)Mr * 32, 32, Nn);
    k_copy_cols<<<g1((long)Mr * 128), 256, 0, stream>>>(g_h,   hcat_g, (long)Mr * 128, 128, 512, 0);
    k_copy_cols<<<g1((long)Mr * 32 ), 256, 0, stream>>>(g_lab, hcat_g, (long)Mr * 32,  32,  512, 384);
    for (int l = 0; l < 2; ++l) {
        const int b = 14 + 12 * l; // gcn_X.{W,b}, gcn_l.{W,b}, ln_X.{b,g}, ln_l.{b,g}, up_X.{W,b}, up_l.{W,b}
        k_copy_cols<<<g1((long)Mr * 128), 256, 0, stream>>>(g_h,   g_cat0, (long)Mr * 128, 128, 160, 0);
        k_copy_cols<<<g1((long)Mr * 32 ), 256, 0, stream>>>(g_lab, g_cat0, (long)Mr * 32,  32,  160, 128);
        gemmF(g_cat0, P(b + 0), nullptr, nullptr, g_t128, Mr, 128, 160, false); // cat@W (bias after aggr)
        gemmAgg(Anorm, g_t128, P(b + 1), g_hx, 128);                            // hx = A@(..)+b
        gemmF(g_lab, P(b + 2), nullptr, nullptr, g_t32, Mr, 32, 32, false);
        gemmAgg(Anorm, g_t32, P(b + 3), g_lb, 32);                              // lb = A@(..)+b
        k_copy_cols<<<g1((long)Mr * 128), 256, 0, stream>>>(g_hx, g_cat2, (long)Mr * 128, 128, 192, 0);
        k_copy_cols<<<g1((long)Mr * 32 ), 256, 0, stream>>>(g_lb, g_cat2, (long)Mr * 32,  32,  192, 128);
        k_copy_cols<<<g1((long)Mr * 32 ), 256, 0, stream>>>(g_ye, g_cat2, (long)Mr * 32,  32,  192, 160);
        gemmF(g_cat2, P(b + 8), P(b + 9), nullptr, g_pre, Mr, 128, 192, false);
        ln_relu_mask<<<lnGrid, 256, 0, stream>>>(g_pre, P(b + 5), P(b + 4), mask, g_h, Mr, 128);
        gemmF(g_lb, P(b + 10), P(b + 11), nullptr, g_lpre, Mr, 32, 32, false);
        ln_relu_mask<<<lnGrid, 256, 0, stream>>>(g_lpre, P(b + 7), P(b + 6), mask, g_lab, Mr, 32);
        k_copy_cols<<<g1((long)Mr * 128), 256, 0, stream>>>(g_h,   hcat_g, (long)Mr * 128, 128, 512, 128 * (l + 1));
        k_copy_cols<<<g1((long)Mr * 32 ), 256, 0, stream>>>(g_lab, hcat_g, (long)Mr * 32,  32,  512, 384 + 32 * (l + 1));
    }
    k_copy_cols<<<g1((long)Mr * 32), 256, 0, stream>>>(g_ye, hcat_g, (long)Mr * 32, 32, 512, 480);
    gemmF(hcat_g, P(38), P(39), nullptr, g_eh1, Mr, 512, 512, true);
    gemmF(g_eh1,  P(40), P(41), nullptr, g_eh,  Mr, 128, 512, false);
    gemmF(g_eh,   P(42), P(43), nullptr, g_e2,  Mr, 128, 128, true);
    gemmF(g_e2,   P(44), P(45), mask, out + (long)Mr * 32, Mr, 2, 128, false); // Em, masked

    k_copy<<<dim3(1), 64, 0, stream>>>(Y, out + (long)Mr * 32 + (long)Mr * 2, 64); // y passthrough
}
